// GatedAttention_64261300683305
// MI455X (gfx1250) — matrix operand order repaired and checked by reading, not compiled, not run on hardware
//
#include <hip/hip_runtime.h>
#include <hip/hip_bf16.h>

// ---------------------------------------------------------------------------
// Gated GQA attention block for MI455X (gfx1250), f16 WMMA + f32 accumulate.
// ---------------------------------------------------------------------------

#define HDIM   2048   // hidden size H
#define SLEN   2048   // sequence length S
#define NHEAD  32
#define KVHEAD 8
#define HD     64
#define NQ     2080   // NH*HD + NH
#define NQPAD  2176   // NQ padded to multiple of 128

typedef _Float16 h16;
typedef _Float16 v16h __attribute__((ext_vector_type(16)));
typedef _Float16 v8h  __attribute__((ext_vector_type(8)));
typedef float    v8f  __attribute__((ext_vector_type(8)));
typedef int      v4i  __attribute__((vector_size(16)));

#define AS1 __attribute__((address_space(1)))
#define AS3 __attribute__((address_space(3)))

#if defined(__has_builtin)
#if __has_builtin(__builtin_amdgcn_global_load_async_to_lds_b128)
#define HAVE_ASYNC_LDS 1
#endif
#endif

__device__ __forceinline__ v8f wmma16x16x32(v16h a, v16h b, v8f c) {
    return __builtin_amdgcn_wmma_f32_16x16x32_f16(false, a, false, b,
                                                  (short)0, c, false, false);
}

__device__ __forceinline__ void stage_b128(const h16* gsrc, h16* ldst) {
#if defined(HAVE_ASYNC_LDS)
    __builtin_amdgcn_global_load_async_to_lds_b128(
        (AS1 v4i*)gsrc, (AS3 v4i*)ldst, 0, 0);
#else
    *(v8h*)ldst = *(const v8h*)gsrc;
#endif
}

__device__ __forceinline__ void stage_wait() {
#if defined(HAVE_ASYNC_LDS)
#if __has_builtin(__builtin_amdgcn_s_wait_asynccnt)
    __builtin_amdgcn_s_wait_asynccnt(0);
#else
    asm volatile("s_wait_asynccnt 0" ::: "memory");
#endif
#endif
}

// ---------------------------------------------------------------------------
// Prep: f32 -> f16 copy
// ---------------------------------------------------------------------------
__global__ void k_f32_to_f16(const float* __restrict__ src,
                             h16* __restrict__ dst, int n) {
    int i = blockIdx.x * 256 + threadIdx.x;
    if (i < n) dst[i] = (h16)src[i];
}

// Prep: W[K][N] f32 -> WT[Npad][K] f16 (zero pad rows n >= N)
__global__ void k_w_transpose(const float* __restrict__ W,
                              h16* __restrict__ WT, int K, int N, int Npad) {
    int i = blockIdx.x * 256 + threadIdx.x;
    if (i >= Npad * K) return;
    int n = i / K, k = i - n * K;
    WT[i] = (n < N) ? (h16)W[(size_t)k * N + n] : (h16)0.0f;
}

// ---------------------------------------------------------------------------
// Tiled WMMA GEMM: C[M=SLEN][N] (f32) = A[SLEN][K] (f16) x BT[N][K] (f16).
// 256 threads = 8 waves; block tile 128x128; K staged 64 at a time in LDS
// (async-to-LDS when available); wave tile 32x64 -> 16 WMMAs per K step.
// N, K compile-time so epilogue strides fold into store offsets.
// ---------------------------------------------------------------------------
template<int N, int K>
__launch_bounds__(256)
__global__ void k_gemm(const h16* __restrict__ A, const h16* __restrict__ BT,
                       float* __restrict__ C) {
    __shared__ __align__(16) h16 As[128 * 72];   // 64 K-halves + 8 pad
    __shared__ __align__(16) h16 Bs[128 * 72];
    const int tid  = threadIdx.x;
    const int lane = tid & 31, wave = tid >> 5;
    const int wr = wave & 3, wc = wave >> 2;     // 4x2 wave grid
    const int m0 = blockIdx.y * 128, n0 = blockIdx.x * 128;
    const int lrow = lane & 15, lhi = lane >> 4;

    v8f acc[2][4] = {};

    for (int k0 = 0; k0 < K; k0 += 64) {
        // stage A (128x64) and BT (128x64): 1024 16B chunks each
        #pragma unroll
        for (int i = 0; i < 4; ++i) {
            int c    = tid + i * 256;
            int row  = c >> 3;
            int koff = (c & 7) * 8;
            stage_b128(&A[(size_t)(m0 + row) * K + k0 + koff],
                       &As[row * 72 + koff]);
            stage_b128(&BT[(size_t)(n0 + row) * K + k0 + koff],
                       &Bs[row * 72 + koff]);
        }
        stage_wait();
        __syncthreads();

        #pragma unroll
        for (int ks = 0; ks < 2; ++ks) {
            v16h a[2], b[4];
            #pragma unroll
            for (int i = 0; i < 2; ++i) {
                int row = wr * 32 + i * 16 + lrow;
                int klo = ks * 32 + lhi * 16;
                ((v8h*)&a[i])[0] = *(const v8h*)&As[row * 72 + klo];
                ((v8h*)&a[i])[1] = *(const v8h*)&As[row * 72 + klo + 8];
            }
            #pragma unroll
            for (int j = 0; j < 4; ++j) {
                int nr  = wc * 64 + j * 16 + lrow;
                int klo = ks * 32 + lhi * 16;
                ((v8h*)&b[j])[0] = *(const v8h*)&Bs[nr * 72 + klo];
                ((v8h*)&b[j])[1] = *(const v8h*)&Bs[nr * 72 + klo + 8];
            }
            #pragma unroll
            for (int i = 0; i < 2; ++i)
                #pragma unroll
                for (int j = 0; j < 4; ++j)
                    acc[i][j] = wmma16x16x32(a[i], b[j], acc[i][j]);
        }
        __syncthreads();
    }

    #pragma unroll
    for (int i = 0; i < 2; ++i)
        #pragma unroll
        for (int j = 0; j < 4; ++j) {
            int row0 = m0 + wr * 32 + i * 16 + 8 * lhi;
            int col  = n0 + wc * 64 + j * 16 + lrow;
            float* cp = C + (size_t)row0 * N + col;
            #pragma unroll
            for (int r = 0; r < 8; ++r)
                cp[(size_t)r * N] = acc[i][j][r];
        }
}

// ---------------------------------------------------------------------------
// RMSNorm + RoPE for q and k. One wave per (s, head); lane i handles the
// rotation pair (d = i, d = i + 32). Output f16 into Qbuf[NH][S][HD] /
// Kbuf[KVH][S][HD].
// ---------------------------------------------------------------------------
__launch_bounds__(128)
__global__ void k_norm_rope(const float* __restrict__ QG,   // [S][NQPAD]
                            const float* __restrict__ Kp,   // [S][KVH*HD]
                            const float* __restrict__ gq,
                            const float* __restrict__ gk,
                            h16* __restrict__ Qbuf,
                            h16* __restrict__ Kbuf) {
    int wid  = blockIdx.x * 4 + (threadIdx.x >> 5);
    int lane = threadIdx.x & 31;

    const float* src;
    const float* gw;
    h16* dst;
    int pos;
    if (wid < SLEN * NHEAD) {
        int s = wid / NHEAD, h = wid - s * NHEAD;
        int kvh = h >> 2, g = h & 3;
        src = QG + (size_t)s * NQPAD + kvh * 260 + g * 64;
        gw  = gq;
        dst = Qbuf + ((size_t)h * SLEN + s) * HD;
        pos = s;
    } else {
        int w2 = wid - SLEN * NHEAD;
        int s = w2 / KVHEAD, kvh = w2 - s * KVHEAD;
        src = Kp + (size_t)s * (KVHEAD * HD) + kvh * HD;
        gw  = gk;
        dst = Kbuf + ((size_t)kvh * SLEN + s) * HD;
        pos = s;
    }

    float x1 = src[lane];
    float x2 = src[lane + 32];
    float ss = x1 * x1 + x2 * x2;
    #pragma unroll
    for (int m = 16; m >= 1; m >>= 1) ss += __shfl_xor(ss, m, 32);
    float inv = rsqrtf(ss * (1.0f / 64.0f) + 1e-6f);
    float xn1 = x1 * inv * gw[lane];
    float xn2 = x2 * inv * gw[lane + 32];

    float fr = (float)pos * __powf(1.0e6f, -(float)lane * (1.0f / 32.0f));
    float sn, cs;
    __sincosf(fr, &sn, &cs);
    dst[lane]      = (h16)(xn1 * cs - xn2 * sn);
    dst[lane + 32] = (h16)(xn2 * cs + xn1 * sn);
}

// Gate: sigmoid of the per-head gate logits from the fused q/gate projection.
__global__ void k_gate(const float* __restrict__ QG, float* __restrict__ gateS) {
    int i = blockIdx.x * 256 + threadIdx.x;
    if (i >= SLEN * NHEAD) return;
    int s = i / NHEAD, h = i - s * NHEAD;
    int kvh = h >> 2, g = h & 3;
    float v = QG[(size_t)s * NQPAD + kvh * 260 + 256 + g];
    gateS[i] = 1.0f / (1.0f + __expf(-v));
}

// V: f32 [S][KVH*HD] -> f16 transposed VT[KVH][HD][S] (contiguous along key).
__global__ void k_v_transpose(const float* __restrict__ Vp, h16* __restrict__ VT) {
    int i = blockIdx.x * 256 + threadIdx.x;
    if (i >= SLEN * KVHEAD * HD) return;
    int s = i / (KVHEAD * HD);
    int c = i - s * (KVHEAD * HD);
    int kvh = c / HD, d = c - kvh * HD;
    VT[((size_t)kvh * HD + d) * SLEN + s] = (h16)Vp[i];
}

// ---------------------------------------------------------------------------
// Causal flash attention with GQA + sigmoid gating.
// One wave per (head, 16-query block); 32 keys per step:
//   scores = Q(16x64) K^T -> two 16x16 f32 C tiles (4 WMMAs),
//   online softmax (half-wave shfl reductions),
//   probs -> LDS -> A fragment -> PV (4 WMMAs into 16x64 f32 accum).
// ---------------------------------------------------------------------------
__launch_bounds__(128)
__global__ void k_attn(const h16* __restrict__ Qbuf,   // [NH][S][HD]
                       const h16* __restrict__ Kbuf,   // [KVH][S][HD]
                       const h16* __restrict__ VT,     // [KVH][HD][S]
                       const float* __restrict__ gateS,// [S][NH]
                       h16* __restrict__ attn16) {     // [S][NH*HD]
    __shared__ __align__(16) h16 P[4][16 * 40];
    const int wave = threadIdx.x >> 5, lane = threadIdx.x & 31;
    const int wid = blockIdx.x * 4 + wave;
    const int h   = wid >> 7;          // 128 query blocks per head
    const int qb  = wid & 127;
    const int q0  = qb * 16;
    const int kvh = h >> 2;
    const int lrow = lane & 15, lhi = lane >> 4;

    // Q fragments (d 0..31 and 32..63)
    const h16* Qb = Qbuf + ((size_t)h * SLEN + q0) * HD;
    v16h aq[2];
    #pragma unroll
    for (int i = 0; i < 2; ++i) {
        const h16* p = Qb + (size_t)lrow * HD + i * 32 + lhi * 16;
        ((v8h*)&aq[i])[0] = *(const v8h*)p;
        ((v8h*)&aq[i])[1] = *(const v8h*)(p + 8);
    }

    v8f o[4] = {};
    float mrun[8], lrun[8];
    #pragma unroll
    for (int r = 0; r < 8; ++r) { mrun[r] = -3.0e38f; lrun[r] = 0.0f; }

    const h16* Kb = Kbuf + (size_t)kvh * SLEN * HD;
    const h16* Vb = VT + (size_t)kvh * HD * SLEN;
    h16* Pl = &P[wave][0];

    for (int j0 = 0; j0 <= q0 + 15; j0 += 32) {
        // ---- scores: two 16x16 tiles over 32 keys ----
        v8f sc[2] = {};
        #pragma unroll
        for (int t = 0; t < 2; ++t) {
            const h16* kp = Kb + (size_t)(j0 + t * 16 + lrow) * HD + lhi * 16;
            v16h b0, b1;
            ((v8h*)&b0)[0] = *(const v8h*)kp;
            ((v8h*)&b0)[1] = *(const v8h*)(kp + 8);
            ((v8h*)&b1)[0] = *(const v8h*)(kp + 32);
            ((v8h*)&b1)[1] = *(const v8h*)(kp + 40);
            sc[t] = wmma16x16x32(aq[0], b0, sc[t]);
            sc[t] = wmma16x16x32(aq[1], b1, sc[t]);
        }

        // ---- online softmax update (per accumulator row) ----
        #pragma unroll
        for (int r = 0; r < 8; ++r) {
            int qrow = q0 + r + 8 * lhi;
            float s0 = sc[0][r] * 0.125f;
            float s1 = sc[1][r] * 0.125f;
            if (j0 + lrow > qrow)      s0 = -3.0e38f;
            if (j0 + 16 + lrow > qrow) s1 = -3.0e38f;
            float mx = fmaxf(s0, s1);
            #pragma unroll
            for (int m = 8; m >= 1; m >>= 1) mx = fmaxf(mx, __shfl_xor(mx, m, 32));
            float mn   = fmaxf(mrun[r], mx);
            float corr = __expf(mrun[r] - mn);
            float p0 = __expf(s0 - mn);
            float p1 = __expf(s1 - mn);
            float rs = p0 + p1;
            #pragma unroll
            for (int m = 8; m >= 1; m >>= 1) rs += __shfl_xor(rs, m, 32);
            lrun[r] = lrun[r] * corr + rs;
            mrun[r] = mn;
            o[0][r] *= corr; o[1][r] *= corr; o[2][r] *= corr; o[3][r] *= corr;
            int prow = r + 8 * lhi;
            Pl[prow * 40 + lrow]      = (h16)p0;
            Pl[prow * 40 + 16 + lrow] = (h16)p1;
        }

        // ---- re-fragment probs (C layout -> A layout via per-wave LDS) ----
        v16h pa;
        {
            const h16* pp = Pl + lrow * 40 + lhi * 16;
            ((v8h*)&pa)[0] = *(const v8h*)pp;
            ((v8h*)&pa)[1] = *(const v8h*)(pp + 8);
        }

        // ---- PV: accumulate into 16x64 output ----
        #pragma unroll
        for (int dt = 0; dt < 4; ++dt) {
            const h16* vp = Vb + (size_t)(dt * 16 + lrow) * SLEN + j0 + lhi * 16;
            v16h bv;
            ((v8h*)&bv)[0] = *(const v8h*)vp;
            ((v8h*)&bv)[1] = *(const v8h*)(vp + 8);
            o[dt] = wmma16x16x32(pa, bv, o[dt]);
        }
    }

    // ---- epilogue: 1/l, sigmoid gate, store f16 [S][NH*HD] ----
    #pragma unroll
    for (int r = 0; r < 8; ++r) {
        int qrow = q0 + r + 8 * lhi;
        float gv   = gateS[(size_t)qrow * NHEAD + h];
        float invl = 1.0f / lrun[r];
        #pragma unroll
        for (int dt = 0; dt < 4; ++dt) {
            attn16[(size_t)qrow * (NHEAD * HD) + h * HD + dt * 16 + lrow] =
                (h16)(o[dt][r] * invl * gv);
        }
    }
}

// ---------------------------------------------------------------------------
// Host-side orchestration
// ---------------------------------------------------------------------------
extern "C" void kernel_launch(void* const* d_in, const int* in_sizes, int n_in,
                              void* d_out, int out_size, void* d_ws, size_t ws_size,
                              hipStream_t stream) {
    const float* X  = (const float*)d_in[0];   // [S][H]
    const float* Wq = (const float*)d_in[1];   // [H][NQ]
    const float* Wk = (const float*)d_in[2];   // [H][KVH*HD]
    const float* Wv = (const float*)d_in[3];   // [H][KVH*HD]
    const float* Wo = (const float*)d_in[4];   // [NH*HD][H]
    const float* gq = (const float*)d_in[5];   // [HD]
    const float* gk = (const float*)d_in[6];   // [HD]
    float* out = (float*)d_out;                // [S][H]

    char* ws = (char*)d_ws;
    size_t off = 0;
    auto carve = [&](size_t bytes) -> void* {
        void* p = ws + off;
        off += (bytes + 255) & ~(size_t)255;
        return p;
    };
    h16*   Xf16 = (h16*)carve((size_t)SLEN * HDIM * 2);
    h16*   WqT  = (h16*)carve((size_t)NQPAD * HDIM * 2);
    h16*   WkT  = (h16*)carve((size_t)KVHEAD * HD * HDIM * 2);
    h16*   WvT  = (h16*)carve((size_t)KVHEAD * HD * HDIM * 2);
    h16*   WoT  = (h16*)carve((size_t)HDIM * NHEAD * HD * 2);
    float* QG   = (float*)carve((size_t)SLEN * NQPAD * 4);
    float* Kp   = (float*)carve((size_t)SLEN * KVHEAD * HD * 4);
    float* Vp   = (float*)carve((size_t)SLEN * KVHEAD * HD * 4);
    h16*   Qbuf = (h16*)carve((size_t)NHEAD * SLEN * HD * 2);
    h16*   Kbuf = (h16*)carve((size_t)KVHEAD * SLEN * HD * 2);
    h16*   VTb  = (h16*)carve((size_t)KVHEAD * HD * SLEN * 2);
    float* gate = (float*)carve((size_t)SLEN * NHEAD * 4);
    h16*   A16  = (h16*)carve((size_t)SLEN * NHEAD * HD * 2);
    (void)ws_size; (void)in_sizes; (void)n_in; (void)out_size;

    // 1. convert / transpose operands to f16
    k_f32_to_f16<<<(SLEN * HDIM + 255) / 256, 256, 0, stream>>>(X, Xf16, SLEN * HDIM);
    k_w_transpose<<<(NQPAD * HDIM + 255) / 256, 256, 0, stream>>>(Wq, WqT, HDIM, NQ, NQPAD);
    k_w_transpose<<<(KVHEAD * HD * HDIM + 255) / 256, 256, 0, stream>>>(Wk, WkT, HDIM, KVHEAD * HD, KVHEAD * HD);
    k_w_transpose<<<(KVHEAD * HD * HDIM + 255) / 256, 256, 0, stream>>>(Wv, WvT, HDIM, KVHEAD * HD, KVHEAD * HD);
    k_w_transpose<<<(HDIM * NHEAD * HD + 255) / 256, 256, 0, stream>>>(Wo, WoT, NHEAD * HD, HDIM, HDIM);

    // 2. projection GEMMs (WMMA, async-LDS staging)
    dim3 blk(256);
    k_gemm<NQPAD, HDIM><<<dim3(NQPAD / 128, SLEN / 128), blk, 0, stream>>>(Xf16, WqT, QG);
    k_gemm<KVHEAD * HD, HDIM><<<dim3((KVHEAD * HD) / 128, SLEN / 128), blk, 0, stream>>>(Xf16, WkT, Kp);
    k_gemm<KVHEAD * HD, HDIM><<<dim3((KVHEAD * HD) / 128, SLEN / 128), blk, 0, stream>>>(Xf16, WvT, Vp);

    // 3. RMSNorm + RoPE, gate sigmoid, V transpose
    k_norm_rope<<<(SLEN * (NHEAD + KVHEAD)) / 4, 128, 0, stream>>>(QG, Kp, gq, gk, Qbuf, Kbuf);
    k_gate<<<(SLEN * NHEAD + 255) / 256, 256, 0, stream>>>(QG, gate);
    k_v_transpose<<<(SLEN * KVHEAD * HD + 255) / 256, 256, 0, stream>>>(Vp, VTb);

    // 4. causal flash attention + gating (WMMA)
    k_attn<<<(NHEAD * (SLEN / 16)) / 4, 128, 0, stream>>>(Qbuf, Kbuf, VTb, gate, A16);

    // 5. output projection (WMMA)
    k_gemm<HDIM, NHEAD * HD><<<dim3(HDIM / 128, SLEN / 128), blk, 0, stream>>>(A16, WoT, out);
}
